// AttnH2O_68169720922790
// MI455X (gfx1250) — matrix operand order repaired and checked by reading, not compiled, not run on hardware
//
#include <hip/hip_runtime.h>

#define KV_HEADS 8
#define HEADS 32
#define HEAD_DIM 128
#define GROUP 4
#define Q_LEN 2048
#define KV_LEN 4096
#define QK_OFF (KV_LEN - Q_LEN)   // causal: kv allowed iff kv <= qi + QK_OFF
#define M_TILE 32                 // queries per block (2 WMMA row-tiles)
#define NTHREADS 512              // 16 waves (wave32)
#define NWAVES 16
#define KV_ELEMS ((size_t)KV_LEN * KV_HEADS * HEAD_DIM)   // 4,194,304

typedef __attribute__((ext_vector_type(16))) _Float16 v16h;
typedef __attribute__((ext_vector_type(8)))  _Float16 v8h;
typedef __attribute__((ext_vector_type(8)))  float    v8f;

static __device__ __forceinline__ v16h make16(v8h lo, v8h hi) {
    v16h r;
    #pragma unroll
    for (int e = 0; e < 8; ++e) { r[e] = lo[e]; r[e + 8] = hi[e]; }
    return r;
}

__global__ void h2o_zero_kernel(float* __restrict__ h2o) {
    int i = blockIdx.x * blockDim.x + threadIdx.x;
    if (i < KV_HEADS * KV_LEN) h2o[i] = 0.0f;
}

// One-time f32 -> f16 conversion of K (same layout). 8 elems/thread.
__global__ __launch_bounds__(512) void cvt_k_kernel(const float* __restrict__ k,
                                                    _Float16* __restrict__ k16) {
    size_t base = ((size_t)blockIdx.x * blockDim.x + threadIdx.x) * 8;
    if (base < KV_ELEMS) {
        v8h o;
        #pragma unroll
        for (int e = 0; e < 8; ++e) o[e] = (_Float16)k[base + e];
        *(v8h*)(k16 + base) = o;
    }
}

// One-time V transpose+convert: vt16[kh][d][kv] = (f16) v[kv][kh*128+d].
__global__ __launch_bounds__(512) void cvt_vt_kernel(const float* __restrict__ v,
                                                     _Float16* __restrict__ vt16) {
    int t    = blockIdx.x * blockDim.x + threadIdx.x;   // 524288 total
    int kv8  = t & (KV_LEN / 8 - 1);                    // kv chunk (512 per row)
    int rest = t >> 9;                                  // kh*128 + d  (1024 rows)
    int d    = rest & (HEAD_DIM - 1);
    int kh   = rest >> 7;
    if (kh < KV_HEADS) {
        const float* src = v + (size_t)(kv8 * 8) * (KV_HEADS * HEAD_DIM)
                             + kh * HEAD_DIM + d;
        v8h o;
        #pragma unroll
        for (int e = 0; e < 8; ++e)
            o[e] = (_Float16)src[(size_t)e * (KV_HEADS * HEAD_DIM)];
        *(v8h*)(vt16 + (size_t)rest * KV_LEN + kv8 * 8) = o;
    }
}

// One block = (query head h, tile of 32 queries), 512 threads = 16 waves.
// Dynamic LDS holds the full 32 x 4096 score/prob tile in f16 (256 KB of the
// WGP's 320 KB) -> exact softmax + exact H2O with a single QK^T and single PV.
template <bool F16CACHE>
__global__ __launch_bounds__(NTHREADS) void attn_h2o_kernel(
    const float*    __restrict__ q,     // [Q_LEN, HEADS*HEAD_DIM]
    const float*    __restrict__ k,     // [KV_LEN, KV_HEADS*HEAD_DIM]
    const float*    __restrict__ v,     // [KV_LEN, KV_HEADS*HEAD_DIM]
    const _Float16* __restrict__ k16,   // [KV_LEN, KV_HEADS*HEAD_DIM] (f16)
    const _Float16* __restrict__ vt16,  // [KV_HEADS, HEAD_DIM, KV_LEN] (f16, V^T)
    float*          __restrict__ out,   // [Q_LEN, HEADS*HEAD_DIM]
    float*          __restrict__ h2o)   // [KV_HEADS, KV_LEN] (pre-zeroed)
{
    extern __shared__ char smem_raw[];
    _Float16* Sh = reinterpret_cast<_Float16*>(smem_raw);   // [M_TILE][KV_LEN]
    __shared__ float invl[M_TILE];

    const int qt   = blockIdx.x;
    const int h    = blockIdx.y;
    const int kh   = h / GROUP;
    const int q0   = qt * M_TILE;
    const int lane = threadIdx.x & 31;
    const int wave = threadIdx.x >> 5;

    const int  mrow   = lane & 15;        // A row / B,C,D column lane index
    const bool hi     = lane >= 16;
    const int  kbaseA = hi ? 8 : 0;       // f16 A-matrix K sub-offset per lane half
    const int  koffB  = hi ? 16 : 0;      // unused; B fragments also use kbaseA offsets

    // ---------------- Phase 0: Q A-fragments for both row tiles --------------
    const float scale = 0.08838834764831845f;   // 1/sqrt(128)
    v16h qfrag[2][4];
    #pragma unroll
    for (int rt = 0; rt < 2; ++rt) {
        const float* qrow = q + (size_t)(q0 + rt * 16 + mrow) * (HEADS * HEAD_DIM)
                              + h * HEAD_DIM;
        #pragma unroll
        for (int db = 0; db < 4; ++db) {
            const float* p0 = qrow + db * 32 + kbaseA;
            #pragma unroll
            for (int e = 0; e < 16; ++e) {
                int d = (e < 8) ? e : (e + 8);          // {0..7, 16..23}
                qfrag[rt][db][e] = (_Float16)(p0[d] * scale);
            }
        }
    }

    // ---------------- Phase 1: S = Q K^T, mask, -> LDS f16 -------------------
    const int kv_hi = q0 + M_TILE - 1 + QK_OFF;   // block's max unmasked kv
    for (int nt = wave; nt < KV_LEN / 16; nt += NWAVES) {
        const int n0 = nt * 16;
        const int kv = n0 + mrow;                 // lane's kv column
        v8f c0 = {}, c1 = {};
        if (n0 <= kv_hi) {
            if constexpr (F16CACHE) {
                const _Float16* krow = k16 + (size_t)kv * (KV_HEADS * HEAD_DIM)
                                           + kh * HEAD_DIM;
                __builtin_prefetch(krow + (size_t)16 * NWAVES * (KV_HEADS * HEAD_DIM),
                                   0, 3);
                // Hoist all 4 d-block fragments: 8 b128 loads issue as one
                // clause, WMMAs retire against decreasing loadcnt (pipelined).
                v8h kb0[4], kb1[4];
                #pragma unroll
                for (int db = 0; db < 4; ++db) {
                    const v8h* p0 = (const v8h*)(krow + db * 32 + kbaseA);
                    kb0[db] = p0[0];
                    kb1[db] = p0[2];
                }
                #pragma unroll
                for (int db = 0; db < 4; ++db) {
                    v16h b = make16(kb0[db], kb1[db]);
                    c0 = __builtin_amdgcn_wmma_f32_16x16x32_f16(
                             false, qfrag[0][db], false, b, (short)0, c0, false, false);
                    c1 = __builtin_amdgcn_wmma_f32_16x16x32_f16(
                             false, qfrag[1][db], false, b, (short)0, c1, false, false);
                }
            } else {
                const float* krow = k + (size_t)kv * (KV_HEADS * HEAD_DIM)
                                      + kh * HEAD_DIM;
                __builtin_prefetch(krow + (size_t)16 * NWAVES * (KV_HEADS * HEAD_DIM),
                                   0, 3);
                #pragma unroll
                for (int db = 0; db < 4; ++db) {
                const float* p0 = krow + db * 32 + kbaseA;
                    v16h b;
                    #pragma unroll
                for (int e = 0; e < 16; ++e) b[e] = (_Float16)p0[(e < 8) ? e : (e + 8)];
                    c0 = __builtin_amdgcn_wmma_f32_16x16x32_f16(
                             false, qfrag[0][db], false, b, (short)0, c0, false, false);
                    c1 = __builtin_amdgcn_wmma_f32_16x16x32_f16(
                             false, qfrag[1][db], false, b, (short)0, c1, false, false);
                }
            }
        }
        #pragma unroll
        for (int vv = 0; vv < 8; ++vv) {
            const int row = vv + (hi ? 8 : 0);    // C/D row within row-tile
            const int qi0 = q0 + row;
            const int qi1 = q0 + 16 + row;
            _Float16 s0 = (kv > qi0 + QK_OFF) ? (_Float16)(-65504.0f) : (_Float16)c0[vv];
            _Float16 s1 = (kv > qi1 + QK_OFF) ? (_Float16)(-65504.0f) : (_Float16)c1[vv];
            Sh[(size_t)row * KV_LEN + kv]        = s0;
            Sh[(size_t)(16 + row) * KV_LEN + kv] = s1;
        }
    }
    __syncthreads();

    // ---------------- Phase 2: exact softmax (vectorized b128 LDS) -----------
    const float LOG2E = 1.4426950408889634f;
    const int   kcap  = (kv_hi + 256) & ~255;     // causal cap, 256-aligned
    #pragma unroll
    for (int rr = 0; rr < 2; ++rr) {
        const int r = wave * 2 + rr;
        _Float16* srow = Sh + (size_t)r * KV_LEN;
        float mx = -3.0e38f;
        for (int k0 = lane * 8; k0 < kcap; k0 += 256) {
            v8h sv = *(const v8h*)(srow + k0);
            #pragma unroll
            for (int e = 0; e < 8; ++e) mx = fmaxf(mx, (float)sv[e]);
        }
        #pragma unroll
        for (int off = 16; off > 0; off >>= 1)
            mx = fmaxf(mx, __shfl_xor(mx, off, 32));
        float sum = 0.0f;
        for (int k0 = lane * 8; k0 < kcap; k0 += 256) {
            v8h sv = *(const v8h*)(srow + k0);
            v8h ev;
            #pragma unroll
            for (int e = 0; e < 8; ++e) {
                float ex = exp2f(((float)sv[e] - mx) * LOG2E);
                sum += ex;
                ev[e] = (_Float16)ex;
            }
            *(v8h*)(srow + k0) = ev;              // in-place: same elements
        }
        #pragma unroll
        for (int off = 16; off > 0; off >>= 1)
            sum += __shfl_xor(sum, off, 32);
        if (lane == 0) invl[r] = 1.0f / sum;
    }
    __syncthreads();

    // ---------------- Phase 3a: H2O column sums (one v8h chunk/thread) -------
    {
        const int k0 = (int)threadIdx.x * 8;      // 512 threads x 8 = 4096
        if (k0 <= kv_hi) {
            float acc[8] = {0.f, 0.f, 0.f, 0.f, 0.f, 0.f, 0.f, 0.f};
            #pragma unroll
            for (int m = 0; m < M_TILE; ++m) {
                v8h pv = *(const v8h*)(Sh + (size_t)m * KV_LEN + k0);
                const float w = invl[m];
                #pragma unroll
                for (int e = 0; e < 8; ++e) acc[e] += (float)pv[e] * w;
            }
            float* h2orow = h2o + (size_t)kh * KV_LEN + k0;
            #pragma unroll
            for (int e = 0; e < 8; ++e)
                atomicAdd(h2orow + e, 0.25f * acc[e]);   // mean over GROUP=4
        }
    }

    // ---------------- Phase 3b: O = P V; wave -> (row-tile, d-chunk) ---------
    {
        const int rt    = wave >> 3;                  // row tile 0/1
        const int dw    = wave & 7;                   // d-chunk [16*dw, 16*dw+16)
        const int dcol  = dw * 16 + mrow;             // B "n" column
        const int kcMax = (q0 + rt * 16 + 15 + QK_OFF) >> 5;
        const _Float16* prow  = Sh + (size_t)(rt * 16 + mrow) * KV_LEN;
        v8f acc = {};
        if constexpr (F16CACHE) {
            const _Float16* vtrow = vt16 + (size_t)(kh * HEAD_DIM + dcol) * KV_LEN;
            const int nIt = kcMax + 1;
            int kc = 0;
            // Unroll by 2 with independent fragment registers: iteration i+1's
            // ds/global loads overlap iteration i's WMMA (acc chain is serial).
            for (; kc + 2 <= nIt; kc += 2) {
                const v8h* pa0 = (const v8h*)(prow + kc * 32 + kbaseA);
                const v8h* pa1 = (const v8h*)(prow + kc * 32 + 32 + kbaseA);
                v8h a00 = pa0[0], a01 = pa0[2];
                v8h a10 = pa1[0], a11 = pa1[2];
                const v8h* pb0 = (const v8h*)(vtrow + kc * 32 + kbaseA);
                const v8h* pb1 = (const v8h*)(vtrow + kc * 32 + 32 + kbaseA);
                v8h b00 = pb0[0], b01 = pb0[2];
                v8h b10 = pb1[0], b11 = pb1[2];
                __builtin_prefetch(vtrow + kc * 32 + 1024, 0, 3);
                acc = __builtin_amdgcn_wmma_f32_16x16x32_f16(
                          false, make16(a00, a01), false, make16(b00, b01),
                          (short)0, acc, false, false);
                acc = __builtin_amdgcn_wmma_f32_16x16x32_f16(
                          false, make16(a10, a11), false, make16(b10, b11),
                          (short)0, acc, false, false);
            }
            if (kc < nIt) {
                const v8h* pa = (const v8h*)(prow + kc * 32 + kbaseA);
                const v8h* pb = (const v8h*)(vtrow + kc * 32 + kbaseA);
                acc = __builtin_amdgcn_wmma_f32_16x16x32_f16(
                          false, make16(pa[0], pa[2]), false, make16(pb[0], pb[2]),
                          (short)0, acc, false, false);
            }
        } else {
            const float* vcol = v + (size_t)kh * HEAD_DIM + dcol;
            for (int kc = 0; kc <= kcMax; ++kc) {
                const v8h* pa = (const v8h*)(prow + kc * 32 + kbaseA);
                v16h a = make16(pa[0], pa[2]);
                const int kvb = kc * 32 + kbaseA;
                __builtin_prefetch(vcol + (size_t)(kvb + 32) * (KV_HEADS * HEAD_DIM),
                                   0, 3);
                v16h b;
                #pragma unroll
                for (int e = 0; e < 16; ++e)
                    b[e] = (_Float16)vcol[(size_t)(kvb + ((e < 8) ? e : (e + 8))) * (KV_HEADS * HEAD_DIM)];
                acc = __builtin_amdgcn_wmma_f32_16x16x32_f16(
                          false, a, false, b, (short)0, acc, false, false);
            }
        }
        #pragma unroll
        for (int vv = 0; vv < 8; ++vv) {
            const int row = rt * 16 + vv + (hi ? 8 : 0);
            out[(size_t)(q0 + row) * (HEADS * HEAD_DIM) + h * HEAD_DIM + dw * 16 + mrow]
                = acc[vv] * invl[row];
        }
    }
}

extern "C" void kernel_launch(void* const* d_in, const int* in_sizes, int n_in,
                              void* d_out, int out_size, void* d_ws, size_t ws_size,
                              hipStream_t stream) {
    const float* q = (const float*)d_in[0];
    const float* k = (const float*)d_in[1];
    const float* v = (const float*)d_in[2];
    float* out = (float*)d_out;
    float* h2o = out + (size_t)Q_LEN * HEADS * HEAD_DIM;

    h2o_zero_kernel<<<(KV_HEADS * KV_LEN + 255) / 256, 256, 0, stream>>>(h2o);

    dim3 grid(Q_LEN / M_TILE, HEADS);
    const size_t smem = (size_t)M_TILE * KV_LEN * sizeof(_Float16);   // 256 KB
    const size_t need = 2 * KV_ELEMS * sizeof(_Float16);              // 16 MB

    if (ws_size >= need) {
        _Float16* k16  = (_Float16*)d_ws;
        _Float16* vt16 = k16 + KV_ELEMS;
        const int nconv = (int)(KV_ELEMS / 8);                        // 524288
        cvt_k_kernel <<<nconv / 512, 512, 0, stream>>>(k, k16);
        cvt_vt_kernel<<<nconv / 512, 512, 0, stream>>>(v, vt16);
        attn_h2o_kernel<true><<<grid, NTHREADS, smem, stream>>>(
            q, k, v, k16, vt16, out, h2o);
    } else {
        attn_h2o_kernel<false><<<grid, NTHREADS, smem, stream>>>(
            q, k, v, nullptr, nullptr, out, h2o);
    }
}
